// DifferentialAttention_66425964200675
// MI455X (gfx1250) — matrix operand order repaired and checked by reading, not compiled, not run on hardware
//
#include <hip/hip_runtime.h>
#include <hip/hip_bf16.h>
#include <math.h>

// ---------------------------------------------------------------------------
// Differential attention, MI455X (gfx1250):
//   - all GEMMs on v_wmma_f32_16x16x32_bf16 (fp32 accumulate)
//   - all strided B-operand tiles staged LDS via Tensor Data Mover
//     (tensor_load_to_lds + s_wait_tensorcnt), fragments built from LDS
// ---------------------------------------------------------------------------

typedef __bf16 bf16_t;
typedef __attribute__((ext_vector_type(16))) __bf16        v16bf;
typedef __attribute__((ext_vector_type(8)))  float         v8f;
typedef __attribute__((ext_vector_type(4)))  unsigned int  v4u;
typedef __attribute__((ext_vector_type(8)))  int           v8i;
typedef __attribute__((ext_vector_type(4)))  int           v4i;

#define B_   8
#define C_   384
#define C3_  1152
#define NH_  8
#define HD_  48
#define HD2_ 24
#define N_   1024

// --- WMMA bf16 fragment index maps (CDNA5 ISA 7.12.2, wave32) ---------------
// A (16x32, MxK): both lane-halves hold rows M=lane&15.
//   element i -> K = i + (i>=8 ? 8 : 0) + 8*(lane>>4)
static __device__ __forceinline__ int a_k(int i, int half) {
  return i + ((i >= 8) ? 8 : 0) + 8 * half;
}
// B (32x16, KxN): lane holds column N=lane&15; element i -> K = i + (i>=8 ? 8 : 0) + 8*(lane>>4)
static __device__ __forceinline__ int b_k(int i, int half) {
  return i + ((i >= 8) ? 8 : 0) + 8 * half;
}
// C/D (16x16 f32): element r of lane l = D[r + 8*(l>>4)][l&15]

static __device__ __forceinline__ v8f wmma_bf16(v16bf a, v16bf b, v8f c) {
  return __builtin_amdgcn_wmma_f32_16x16x32_bf16(false, a, false, b,
                                                 (short)0, c, false, false);
}

static __device__ __forceinline__ unsigned rfl(unsigned x) {
  return (unsigned)__builtin_amdgcn_readfirstlane((int)x);
}

// --- Tensor Data Mover: 2-D tile load, D# per CDNA5 ISA ch.8 -----------------
// gptr     : global address of tile start
// ldsptr   : LDS destination (tile stored contiguously, row-major)
// ds_code  : data_size field (0=1B,1=2B,2=4B,3=8B)
// tile_d0  : tile width  (contiguous dim, data_size units; x4 bytes multiple)
// tile_d1  : tile height (rows)
// tensor_d0/tensor_d1 : tensor extent for OOB checks (units of data_size)
// stride0  : row stride (units of data_size)
static __device__ __forceinline__ void tdm_load_2d(const void* gptr,
                                                   const void* ldsptr,
                                                   int ds_code,
                                                   int tile_d0, int tile_d1,
                                                   unsigned tensor_d0,
                                                   unsigned tensor_d1,
                                                   unsigned stride0) {
  const unsigned long long ga = (unsigned long long)(size_t)gptr;
  const unsigned lds_off = (unsigned)(size_t)ldsptr;   // addr[31:0] = LDS offset

  v4u g0;
  g0[0] = rfl(1u);                                        // count=1, user mode
  g0[1] = rfl(lds_off);                                   // lds_addr
  g0[2] = rfl((unsigned)(ga & 0xFFFFFFFFu));              // global_addr lo
  g0[3] = rfl((unsigned)((ga >> 32) & 0x01FFFFFFu) | (2u << 30)); // hi | type=2

  v8i g1;
  g1[0] = (int)rfl(((unsigned)ds_code & 3u) << 16);       // wg_mask=0, data_size
  g1[1] = (int)rfl((tensor_d0 & 0xFFFFu) << 16);          // tdim0[15:0]
  g1[2] = (int)rfl(((tensor_d0 >> 16) & 0xFFFFu) | ((tensor_d1 & 0xFFFFu) << 16));
  g1[3] = (int)rfl(((tensor_d1 >> 16) & 0xFFFFu) | ((unsigned)tile_d0 << 16));
  g1[4] = (int)rfl((unsigned)tile_d1 & 0xFFFFu);          // tile_dim1, tile_dim2=0
  g1[5] = (int)rfl(stride0);                              // dim0_stride lo32
  g1[6] = 0;                                              // stride hi, dim1_stride=0
  g1[7] = 0;

  v4i z4 = {0, 0, 0, 0};
#if defined(__clang_major__) && __clang_major__ >= 23
  v8i z8 = {0, 0, 0, 0, 0, 0, 0, 0};
  __builtin_amdgcn_tensor_load_to_lds(g0, g1, z4, z4, z8, 0);
#else
  __builtin_amdgcn_tensor_load_to_lds(g0, g1, z4, z4, 0);
#endif
}

static __device__ __forceinline__ void tdm_wait() {
  __builtin_amdgcn_s_wait_tensorcnt((short)0);
}

// ===========================================================================
// K1: qkv[b,o,n] = sum_c qkv_w[o,c] * x[b,c,n]   (M=1152, K=384, N=1024)
// one wave -> one 16x16 tile; x-tile (32x16 f32) TDM-staged once per block.
// ===========================================================================
__global__ void k_qkv_gemm(const float* __restrict__ x,
                           const float* __restrict__ w,
                           bf16_t* __restrict__ qkv) {
  __shared__ __align__(16) float sXf[32 * 16];   // staged x tile [c_local][n]

  const int lane = threadIdx.x & 31;
  const int wave = threadIdx.x >> 5;
  const int n0   = blockIdx.x * 16;
  const int o0   = (blockIdx.y * 8 + wave) * 16;
  const int b    = blockIdx.z;
  const int col  = lane & 15;
  const int half = lane >> 4;

  const float* xb   = x + (size_t)b * C_ * N_;
  const float* wrow = w + (size_t)(o0 + col) * C_;
  v8f acc = {0.f, 0.f, 0.f, 0.f, 0.f, 0.f, 0.f, 0.f};

  for (int kk = 0; kk < C_; kk += 32) {
    __syncthreads();                       // previous tile fully consumed
    if (wave == 0) {
      tdm_load_2d(xb + (size_t)kk * N_ + n0, sXf, /*4B*/2,
                  16, 32, N_, 32, N_);
      tdm_wait();
    }
    __syncthreads();                       // tile visible to all waves

    // A fragment: two contiguous 8-float spans of one weight row (b128 loads)
    const float4* wr = (const float4*)(wrow + kk);
    const float4 w0 = wr[2 * half + 0];
    const float4 w1 = wr[2 * half + 1];
    const float4 w2 = wr[4 + 2 * half + 0];
    const float4 w3 = wr[4 + 2 * half + 1];
    v16bf a;
    a[0] = (bf16_t)w0.x;  a[1] = (bf16_t)w0.y;  a[2]  = (bf16_t)w0.z;  a[3]  = (bf16_t)w0.w;
    a[4] = (bf16_t)w1.x;  a[5] = (bf16_t)w1.y;  a[6]  = (bf16_t)w1.z;  a[7]  = (bf16_t)w1.w;
    a[8] = (bf16_t)w2.x;  a[9] = (bf16_t)w2.y;  a[10] = (bf16_t)w2.z;  a[11] = (bf16_t)w2.w;
    a[12] = (bf16_t)w3.x; a[13] = (bf16_t)w3.y; a[14] = (bf16_t)w3.z;  a[15] = (bf16_t)w3.w;

    v16bf bb;
#pragma unroll
    for (int i = 0; i < 16; ++i)
      bb[i] = (bf16_t)sXf[b_k(i, half) * 16 + col];

    acc = wmma_bf16(a, bb, acc);
  }

  bf16_t* dst = qkv + (size_t)b * C3_ * N_;
#pragma unroll
  for (int r = 0; r < 8; ++r) {
    const int o = o0 + r + 8 * half;
    dst[(size_t)o * N_ + n0 + col] = (bf16_t)acc[r];
  }
}

// ===========================================================================
// K2: GroupNorm stats over q2 (groups g cover heads 4g..4g+3)
// ===========================================================================
__global__ void k_gn_stats(const bf16_t* __restrict__ qkv,
                           float* __restrict__ stats) {
  const int b = blockIdx.x >> 1;
  const int g = blockIdx.x & 1;
  const int tid = threadIdx.x;
  __shared__ float rs[256], rs2[256];

  const int total = 4 * HD2_ * N_;
  float s = 0.f, s2 = 0.f;
  for (int idx = tid; idx < total; idx += 256) {
    const int ch  = idx >> 10;
    const int pos = idx & 1023;
    const int h   = g * 4 + ch / HD2_;
    const int d   = ch % HD2_;
    const int o   = h * HD_ + HD2_ + d;
    const float v = (float)qkv[((size_t)b * C3_ + o) * N_ + pos];
    s += v; s2 += v * v;
  }
  rs[tid] = s; rs2[tid] = s2;
  __syncthreads();
  for (int off = 128; off > 0; off >>= 1) {
    if (tid < off) { rs[tid] += rs[tid + off]; rs2[tid] += rs2[tid + off]; }
    __syncthreads();
  }
  if (tid == 0) {
    const float inv  = 1.f / (float)total;
    const float mean = rs[0] * inv;
    const float var  = rs2[0] * inv - mean * mean;
    stats[(b * 2 + g) * 2 + 0] = mean;
    stats[(b * 2 + g) * 2 + 1] = rsqrtf(var + 1e-5f);
  }
}

// ===========================================================================
// K3: fused differential attention for one (b, h, 16 query rows).
// Q/K/V tiles TDM-staged to LDS (K-dim rows 24..31 pre-zeroed once).
// Dynamic LDS map (bytes):
//   0      sS   f32 [16][1024]   65536
//   65536  sP1  bf16[16][1024]   32768
//   98304  sO   f32 [16][48]      3072
//   101376 sR   f32 [16][16]      1024
//   102400 sQ   bf16[32][16]      1024   (shared, wave0-staged)
//   103424 sK   bf16[8][32][16]   8192   (per-wave slots)
//   111616 sV   bf16[8][48][32]  24576   (per-wave slots)
//   total 136192
// ===========================================================================
__global__ void k_attn(const bf16_t* __restrict__ qkv,
                       const float* __restrict__ stats,
                       const float* __restrict__ lam,
                       const float* __restrict__ gnw,
                       const float* __restrict__ gnb,
                       bf16_t* __restrict__ aout) {
  extern __shared__ char smem[];
  float*  sS  = (float*)smem;
  bf16_t* sP1 = (bf16_t*)(smem + 65536);
  float*  sO  = (float*)(smem + 98304);
  float*  sR  = (float*)(smem + 101376);
  bf16_t* sQ  = (bf16_t*)(smem + 102400);

  const int n0   = blockIdx.x * 16;
  const int h    = blockIdx.y;
  const int b    = blockIdx.z;
  const int tid  = threadIdx.x;
  const int lane = tid & 31, wave = tid >> 5;
  const int col  = lane & 15, half = lane >> 4;

  bf16_t* sK = (bf16_t*)(smem + 103424 + wave * 1024);   // [32][16]
  bf16_t* sV = (bf16_t*)(smem + 111616 + wave * 3072);   // [48][32]

  const bf16_t* base = qkv + (size_t)b * C3_ * N_;
  const float scale = rsqrtf((float)HD_);
  const float lamh  = lam[h];
  const int   g     = h >> 2;
  const float mean  = stats[(b * 2 + g) * 2 + 0];
  const float rstd  = stats[(b * 2 + g) * 2 + 1];

  // zero out-accumulator + zero-pad rows 24..31 of sQ and this wave's sK
  for (int i = tid; i < 16 * 48; i += 256) sO[i] = 0.f;
  if (tid < 64) ((unsigned*)sQ)[192 + tid] = 0u;
  ((unsigned*)sK)[192 + lane] = 0u;
  ((unsigned*)sK)[224 + lane] = 0u;

  // -------- branch 1: S1 = (q1^T k1) * scale --------
  if (wave == 0) {
    tdm_load_2d(base + (size_t)(h * HD_) * N_ + n0, sQ, /*2B*/1,
                16, HD2_, N_, HD2_, N_);
    tdm_wait();
  }
  __syncthreads();

  v16bf aq;
#pragma unroll
  for (int i = 0; i < 16; ++i)
    aq[i] = sQ[a_k(i, half) * 16 + col];

  for (int mt = wave; mt < 64; mt += 8) {
    tdm_load_2d(base + (size_t)(C_ + h * HD_) * N_ + mt * 16, sK, /*2B*/1,
                16, HD2_, N_, HD2_, N_);
    tdm_wait();
    v16bf bk;
#pragma unroll
    for (int i = 0; i < 16; ++i)
      bk[i] = sK[b_k(i, half) * 16 + col];
    v8f acc = {0.f, 0.f, 0.f, 0.f, 0.f, 0.f, 0.f, 0.f};
    acc = wmma_bf16(aq, bk, acc);
#pragma unroll
    for (int r = 0; r < 8; ++r)
      sS[(r + 8 * half) * 1024 + mt * 16 + col] = acc[r] * scale;
  }
  __syncthreads();

  // softmax rows of sS -> sP1 (bf16). 16 threads per row.
  {
    const int row = tid >> 4, sub = tid & 15;
    float mx = -1e30f;
    for (int c2 = sub; c2 < 1024; c2 += 16) mx = fmaxf(mx, sS[row * 1024 + c2]);
    sR[row * 16 + sub] = mx;
    __syncthreads();
    mx = -1e30f;
#pragma unroll
    for (int j = 0; j < 16; ++j) mx = fmaxf(mx, sR[row * 16 + j]);
    __syncthreads();
    float ls = 0.f;
    for (int c2 = sub; c2 < 1024; c2 += 16) {
      const float e = __expf(sS[row * 1024 + c2] - mx);
      sS[row * 1024 + c2] = e;
      ls += e;
    }
    sR[row * 16 + sub] = ls;
    __syncthreads();
    float tot = 0.f;
#pragma unroll
    for (int j = 0; j < 16; ++j) tot += sR[row * 16 + j];
    const float inv = 1.f / tot;
    for (int c2 = sub; c2 < 1024; c2 += 16)
      sP1[row * 1024 + c2] = (bf16_t)(sS[row * 1024 + c2] * inv);
  }
  __syncthreads();

  // -------- branch 2: S2 = (q2n^T k2) * scale * lambda --------
  if (wave == 0) {
    tdm_load_2d(base + (size_t)(h * HD_ + HD2_) * N_ + n0, sQ, /*2B*/1,
                16, HD2_, N_, HD2_, N_);
    tdm_wait();
  }
  __syncthreads();
#pragma unroll
  for (int i = 0; i < 16; ++i) {
    const int d = a_k(i, half);
    if (d < HD2_) {
      float v = (float)sQ[d * 16 + col];
      v = (v - mean) * rstd * gnw[h * HD2_ + d] + gnb[h * HD2_ + d];
      aq[i] = (bf16_t)v;
    } else {
      aq[i] = (bf16_t)0.f;
    }
  }
  for (int mt = wave; mt < 64; mt += 8) {
    tdm_load_2d(base + (size_t)(C_ + h * HD_ + HD2_) * N_ + mt * 16, sK, /*2B*/1,
                16, HD2_, N_, HD2_, N_);
    tdm_wait();
    v16bf bk;
#pragma unroll
    for (int i = 0; i < 16; ++i)
      bk[i] = sK[b_k(i, half) * 16 + col];
    v8f acc = {0.f, 0.f, 0.f, 0.f, 0.f, 0.f, 0.f, 0.f};
    acc = wmma_bf16(aq, bk, acc);
#pragma unroll
    for (int r = 0; r < 8; ++r)
      sS[(r + 8 * half) * 1024 + mt * 16 + col] = acc[r] * scale * lamh;
  }
  __syncthreads();

  // softmax rows of sS in place (f32)
  {
    const int row = tid >> 4, sub = tid & 15;
    float mx = -1e30f;
    for (int c2 = sub; c2 < 1024; c2 += 16) mx = fmaxf(mx, sS[row * 1024 + c2]);
    sR[row * 16 + sub] = mx;
    __syncthreads();
    mx = -1e30f;
#pragma unroll
    for (int j = 0; j < 16; ++j) mx = fmaxf(mx, sR[row * 16 + j]);
    __syncthreads();
    float ls = 0.f;
    for (int c2 = sub; c2 < 1024; c2 += 16) {
      const float e = __expf(sS[row * 1024 + c2] - mx);
      sS[row * 1024 + c2] = e;
      ls += e;
    }
    sR[row * 16 + sub] = ls;
    __syncthreads();
    float tot = 0.f;
#pragma unroll
    for (int j = 0; j < 16; ++j) tot += sR[row * 16 + j];
    const float inv = 1.f / tot;
    for (int c2 = sub; c2 < 1024; c2 += 16) sS[row * 1024 + c2] *= inv;
  }
  __syncthreads();

  // -------- out = (P1 - P2) @ V^T;  V tile (48 x 32) TDM-staged per chunk ----
  v8f oacc[3];
#pragma unroll
  for (int dt = 0; dt < 3; ++dt)
    oacc[dt] = (v8f){0.f, 0.f, 0.f, 0.f, 0.f, 0.f, 0.f, 0.f};

  for (int ck = wave; ck < 32; ck += 8) {   // 32 key-chunks of 32
    tdm_load_2d(base + (size_t)(2 * C_ + h * HD_) * N_ + ck * 32, sV, /*2B*/1,
                32, HD_, N_, HD_, N_);
    tdm_wait();
    v16bf ap;
#pragma unroll
    for (int i = 0; i < 16; ++i) {
      const int m = ck * 32 + a_k(i, half);
      ap[i] = (bf16_t)((float)sP1[col * 1024 + m] - sS[col * 1024 + m]);
    }
#pragma unroll
    for (int dt = 0; dt < 3; ++dt) {
      v16bf bv;
#pragma unroll
      for (int i = 0; i < 16; ++i)
        bv[i] = sV[(dt * 16 + col) * 32 + b_k(i, half)];
      oacc[dt] = wmma_bf16(ap, bv, oacc[dt]);
    }
  }
  // cross-wave reduction (ds_add_f32)
#pragma unroll
  for (int dt = 0; dt < 3; ++dt)
#pragma unroll
    for (int r = 0; r < 8; ++r)
      atomicAdd(&sO[(r + 8 * half) * 48 + dt * 16 + col], oacc[dt][r]);
  __syncthreads();

  // write out in the reference's concat(n)->reshape(C,N) channel layout
  for (int i = tid; i < 16 * 48; i += 256) {
    const int nl = i / 48, d = i % 48;
    const int n  = n0 + nl;
    const int n2 = (d < HD2_) ? n : (N_ + n);
    const int dd = (d < HD2_) ? d : (d - HD2_);
    const int flat = h * (2 * N_ * HD2_) + n2 * HD2_ + dd;
    const int c2 = flat >> 10, pos = flat & 1023;
    aout[((size_t)b * C_ + c2) * N_ + pos] = (bf16_t)sO[i];
  }
}

// ===========================================================================
// K4: out[b,o,hw] = sum_c proj_w[o,c] * attn[b,c,hw] + proj_b[o]  (f32 out)
// B tile (32x16 bf16) TDM-staged once per block per k-step.
// ===========================================================================
__global__ void k_proj(const float* __restrict__ w,
                       const float* __restrict__ bias,
                       const bf16_t* __restrict__ ain,
                       float* __restrict__ out) {
  __shared__ __align__(16) bf16_t sB[32 * 16];

  const int lane = threadIdx.x & 31;
  const int wave = threadIdx.x >> 5;
  const int n0   = blockIdx.x * 16;
  const int o0   = (blockIdx.y * 8 + wave) * 16;
  const int b    = blockIdx.z;
  const int col  = lane & 15;
  const int half = lane >> 4;

  const bf16_t* ab   = ain + (size_t)b * C_ * N_;
  const float*  wrow = w + (size_t)(o0 + col) * C_;
  v8f acc = {0.f, 0.f, 0.f, 0.f, 0.f, 0.f, 0.f, 0.f};

  for (int kk = 0; kk < C_; kk += 32) {
    __syncthreads();
    if (wave == 0) {
      tdm_load_2d(ab + (size_t)kk * N_ + n0, sB, /*2B*/1,
                  16, 32, N_, 32, N_);
      tdm_wait();
    }
    __syncthreads();

    const float4* wr = (const float4*)(wrow + kk);
    const float4 w0 = wr[2 * half + 0];
    const float4 w1 = wr[2 * half + 1];
    const float4 w2 = wr[4 + 2 * half + 0];
    const float4 w3 = wr[4 + 2 * half + 1];
    v16bf a;
    a[0] = (bf16_t)w0.x;  a[1] = (bf16_t)w0.y;  a[2]  = (bf16_t)w0.z;  a[3]  = (bf16_t)w0.w;
    a[4] = (bf16_t)w1.x;  a[5] = (bf16_t)w1.y;  a[6]  = (bf16_t)w1.z;  a[7]  = (bf16_t)w1.w;
    a[8] = (bf16_t)w2.x;  a[9] = (bf16_t)w2.y;  a[10] = (bf16_t)w2.z;  a[11] = (bf16_t)w2.w;
    a[12] = (bf16_t)w3.x; a[13] = (bf16_t)w3.y; a[14] = (bf16_t)w3.z;  a[15] = (bf16_t)w3.w;

    v16bf bb;
#pragma unroll
    for (int i = 0; i < 16; ++i)
      bb[i] = sB[b_k(i, half) * 16 + col];

    acc = wmma_bf16(a, bb, acc);
  }
#pragma unroll
  for (int r = 0; r < 8; ++r) {
    const int o = o0 + r + 8 * half;
    out[((size_t)b * C_ + o) * N_ + n0 + col] = acc[r] + bias[o];
  }
}

// ===========================================================================
extern "C" void kernel_launch(void* const* d_in, const int* in_sizes, int n_in,
                              void* d_out, int out_size, void* d_ws, size_t ws_size,
                              hipStream_t stream) {
  const float* x     = (const float*)d_in[0];
  const float* qkv_w = (const float*)d_in[1];
  const float* lam   = (const float*)d_in[2];
  const float* gnw   = (const float*)d_in[3];
  const float* gnb   = (const float*)d_in[4];
  const float* pw    = (const float*)d_in[5];
  const float* pb    = (const float*)d_in[6];
  float* out = (float*)d_out;

  // workspace: qkv bf16 (18.9MB) | stats f32 (pad 256B) | attn bf16 (6.3MB)
  bf16_t* qkv = (bf16_t*)d_ws;
  const size_t QKV_BYTES = (size_t)B_ * C3_ * N_ * sizeof(bf16_t);
  float*  stats = (float*)((char*)d_ws + QKV_BYTES);
  bf16_t* aout  = (bf16_t*)((char*)d_ws + QKV_BYTES + 256);

  k_qkv_gemm<<<dim3(N_ / 16, C3_ / 16 / 8, B_), dim3(256), 0, stream>>>(x, qkv_w, qkv);
  k_gn_stats<<<dim3(B_ * 2), dim3(256), 0, stream>>>(qkv, stats);
  k_attn<<<dim3(N_ / 16, NH_, B_), dim3(256), 136192, stream>>>(qkv, stats, lam, gnw, gnb, aout);
  k_proj<<<dim3(N_ / 16, C_ / 16 / 8, B_), dim3(256), 0, stream>>>(pw, pb, aout, out);
}
